// Attention_7318624272435
// MI455X (gfx1250) — matrix operand order repaired and checked by reading, not compiled, not run on hardware
//
#include <hip/hip_runtime.h>
#include <hip/hip_bf16.h>

typedef __attribute__((ext_vector_type(16))) __bf16 v16bf;
typedef __attribute__((ext_vector_type(8)))  float  v8f;

#define EMBED 1024
#define SEQ   2048
#define BATCH 8
#define FOUT  3072   // 3*EMBED

union AFrag { v16bf v; uint4 u[2]; };
union BFrag { v16bf v; uint4 u[2]; };

__device__ inline v8f wmma_bf16(v16bf a, v16bf b, v8f c) {
    return __builtin_amdgcn_wmma_f32_16x16x32_bf16(
        /*neg_a=*/false, a, /*neg_b=*/false, b,
        /*c_mod=*/(short)0, c, /*reuse_a=*/false, /*reuse_b=*/false);
}

__device__ inline void cvt16(const float4* s4, __bf16* d) {
    float4 a = s4[0], b = s4[1], c = s4[2], e = s4[3];
    d[0]=(__bf16)a.x;  d[1]=(__bf16)a.y;  d[2]=(__bf16)a.z;  d[3]=(__bf16)a.w;
    d[4]=(__bf16)b.x;  d[5]=(__bf16)b.y;  d[6]=(__bf16)b.z;  d[7]=(__bf16)b.w;
    d[8]=(__bf16)c.x;  d[9]=(__bf16)c.y;  d[10]=(__bf16)c.z; d[11]=(__bf16)c.w;
    d[12]=(__bf16)e.x; d[13]=(__bf16)e.y; d[14]=(__bf16)e.z; d[15]=(__bf16)e.w;
}

// ---------------------------------------------------------------------------
// Kernel 1: qkv = x @ W^T + b -> bf16 Q, K (row-major) and Vt (e-major).
// 128x128 block tile, K-step 32, double-buffered LDS, 8 waves (4M x 2F),
// epilogue staged through LDS for coalesced b128 stores, region uniform per block.
// ---------------------------------------------------------------------------
#define LDH 40    // stage row stride (halves): 32 data + 8 pad, 80B (16B aligned)
#define LDC 136   // C-stage row stride (halves): 128 data + 8 pad, 272B (16B aligned)

union GemmSmem {
    struct { __bf16 A[2][128 * LDH]; __bf16 B[2][128 * LDH]; } s;  // 40 KB
    __bf16 C[128 * LDC];                                           // 34 KB
};

__global__ __launch_bounds__(256)
void qkv_gemm_kernel(const float* __restrict__ x, const float* __restrict__ W,
                     const float* __restrict__ bias,
                     __bf16* __restrict__ Q, __bf16* __restrict__ K,
                     __bf16* __restrict__ Vt)
{
    __shared__ GemmSmem sm;

    const int tid  = threadIdx.x;
    const int lane = tid & 31;
    const int wave = tid >> 5;
    const int m0 = blockIdx.x * 128;       // 16384/128 = 128
    const int f0 = blockIdx.y * 128;       // 3072/128  = 24
    const int region = blockIdx.y >> 3;    // 0:Q 1:K 2:V (uniform)

    const int wm = wave & 3;               // 4 waves x 32 M-rows
    const int wf = wave >> 2;              // 2 waves x 64 F-cols

    const v8f zero8 = {0.f,0.f,0.f,0.f,0.f,0.f,0.f,0.f};
    v8f acc[2][4];
#pragma unroll
    for (int s = 0; s < 2; ++s)
#pragma unroll
        for (int t = 0; t < 4; ++t) acc[s][t] = zero8;

    const int r = tid >> 1;                // staged row 0..127
    const int h = tid & 1;                 // 16-float half
    const int rA  = lane & 15;
    const int mh  = (lane < 16) ? 0 : 8;
    const int kbA = (lane < 16) ? 0 : 8;
    const int kbB = (lane < 16) ? 0 : 8;

    auto stage = [&](int kk, int p) {
        cvt16((const float4*)(x + (size_t)(m0 + r) * EMBED + kk + h * 16),
              &sm.s.A[p][r * LDH + h * 16]);
        cvt16((const float4*)(W + (size_t)(f0 + r) * EMBED + kk + h * 16),
              &sm.s.B[p][r * LDH + h * 16]);
    };

    stage(0, 0);
    __syncthreads();

    for (int kk = 0; kk < EMBED; kk += 32) {
        const int p = (kk >> 5) & 1;
        if (kk + 32 < EMBED) stage(kk + 32, p ^ 1);   // overlap next-tile loads

        AFrag a[2];
#pragma unroll
        for (int s = 0; s < 2; ++s) {
            const __bf16* ap = &sm.s.A[p][(wm * 32 + s * 16 + rA) * LDH];
            a[s].u[0] = *(const uint4*)(ap + kbA);
            a[s].u[1] = *(const uint4*)(ap + kbA + 16);
        }
        BFrag bb[4];
#pragma unroll
        for (int t = 0; t < 4; ++t) {
            const __bf16* bp = &sm.s.B[p][(wf * 64 + t * 16 + rA) * LDH + kbB];
            bb[t].u[0] = *(const uint4*)(bp);
            bb[t].u[1] = *(const uint4*)(bp + 16);
        }
#pragma unroll
        for (int s = 0; s < 2; ++s)
#pragma unroll
            for (int t = 0; t < 4; ++t)
                acc[s][t] = wmma_bf16(a[s].v, bb[t].v, acc[s][t]);
        __syncthreads();
    }

    // ---- epilogue: +bias, stage tile in LDS, coalesced b128 stores ----
    if (region < 2) {  // Q or K : C layout [m][f]
#pragma unroll
        for (int s = 0; s < 2; ++s)
#pragma unroll
            for (int t = 0; t < 4; ++t) {
                const int fl = wf * 64 + t * 16 + rA;
                const float bv = bias[f0 + fl];
                const int ml = wm * 32 + s * 16 + mh;
#pragma unroll
                for (int j = 0; j < 8; ++j)
                    sm.C[(ml + j) * LDC + fl] = (__bf16)(acc[s][t][j] + bv);
            }
        __syncthreads();
        __bf16* dst = (region == 0) ? Q : K;
        const int fcol0 = f0 - region * EMBED;
        for (int idx = tid; idx < 2048; idx += 256) {
            const int row = idx >> 4, c = (idx & 15) * 8;
            *(uint4*)(dst + (size_t)(m0 + row) * EMBED + fcol0 + c) =
                *(const uint4*)&sm.C[row * LDC + c];
        }
    } else {           // V : C layout [e][m] (transposed for Vt)
#pragma unroll
        for (int s = 0; s < 2; ++s)
#pragma unroll
            for (int t = 0; t < 4; ++t) {
                const int fl = wf * 64 + t * 16 + rA;
                const float bv = bias[f0 + fl];
                const int ml = wm * 32 + s * 16 + mh;
#pragma unroll
                for (int j = 0; j < 8; ++j)
                    sm.C[fl * LDC + (ml + j)] = (__bf16)(acc[s][t][j] + bv);
            }
        __syncthreads();
        const int bb_ = m0 >> 11;              // batch (m0 multiple of 128)
        const int tok0 = m0 & (SEQ - 1);
        const int e0 = f0 - 2 * EMBED;
        __bf16* dst = Vt + ((size_t)bb_ * EMBED + e0) * SEQ + tok0;
        for (int idx = tid; idx < 2048; idx += 256) {
            const int row = idx >> 4, c = (idx & 15) * 8;   // row=e, c=token chunk
            *(uint4*)(dst + (size_t)row * SEQ + c) =
                *(const uint4*)&sm.C[row * LDC + c];
        }
    }
}

// ---------------------------------------------------------------------------
// Kernel 2: flash attention. Block = (batch, 16-query tile), 8 waves,
// key-block Bc=64 (32 WMMAs per 4 barriers). Wave w owns e-cols [w*128,+128).
// ---------------------------------------------------------------------------
#define LDP 72   // Plds row stride (halves): 64 data + 8 pad, 144B (16B aligned)

__global__ __launch_bounds__(256)
void flash_attn_kernel(const __bf16* __restrict__ Q, const __bf16* __restrict__ K,
                       const __bf16* __restrict__ Vt, float* __restrict__ out)
{
    const int b  = blockIdx.y;
    const int q0 = blockIdx.x * 16;
    const __bf16* Qb = Q  + (size_t)b * SEQ * EMBED;
    const __bf16* Kb = K  + (size_t)b * SEQ * EMBED;
    const __bf16* Vb = Vt + (size_t)b * EMBED * SEQ;
    float*        Ob = out + (size_t)b * SEQ * EMBED;

    const int tid  = threadIdx.x;
    const int lane = tid & 31;
    const int wave = tid >> 5;
    const int rA   = lane & 15;
    const int mh   = (lane < 16) ? 0 : 8;

    __shared__ float  Sp[8][4][16][16];   // per-wave partial S tiles (32 KB)
    __shared__ float  Sf[16][64];         // reduced scores (4 KB)
    __shared__ __bf16 Plds[16][LDP];      // exp(P) bf16, A-layout source
    __shared__ float  mrow[16], lrow[16], arow[16];

    if (tid < 16) { mrow[tid] = -1e30f; lrow[tid] = 0.f; arow[tid] = 0.f; }

    // preload Q fragments for this wave's e-slice
    AFrag qf[4];
    {
        const __bf16* qrow = Qb + (size_t)(q0 + rA) * EMBED + wave * 128;
        const int eh = (lane < 16) ? 0 : 8;
#pragma unroll
        for (int f = 0; f < 4; ++f) {
            qf[f].u[0] = *(const uint4*)(qrow + f * 32 + eh);
            qf[f].u[1] = *(const uint4*)(qrow + f * 32 + 16 + eh);
        }
    }

    const v8f zero8 = {0.f,0.f,0.f,0.f,0.f,0.f,0.f,0.f};
    v8f O[8];
#pragma unroll
    for (int s = 0; s < 8; ++s) O[s] = zero8;

    __syncthreads();

    for (int kb = 0; kb < SEQ; kb += 64) {
        // ---- phase 1: partial S = Q*K^T over this wave's e-slice (16 WMMA) ----
        const int ebB = (lane < 16) ? 0 : 8;
#pragma unroll
        for (int t = 0; t < 4; ++t) {
            const int key = kb + t * 16 + rA;
            const __bf16* krow = Kb + (size_t)key * EMBED + wave * 128;
            v8f c = zero8;
#pragma unroll
            for (int f = 0; f < 4; ++f) {
                BFrag kf;
                const __bf16* p = krow + f * 32 + ebB;
                kf.u[0] = *(const uint4*)(p);
                kf.u[1] = *(const uint4*)(p + 16);
                c = wmma_bf16(qf[f].v, kf.v, c);
            }
#pragma unroll
            for (int j = 0; j < 8; ++j) Sp[wave][t][mh + j][rA] = c[j];
        }
        __syncthreads();

        // ---- phase 2: reduce 8 partials, apply 1/sqrt(1024) ----
        for (int i = tid; i < 1024; i += 256) {
            const int m = i >> 6, n = i & 63;
            float s = 0.f;
#pragma unroll
            for (int w = 0; w < 8; ++w) s += Sp[w][n >> 4][m][n & 15];
            Sf[m][n] = s * 0.03125f;
        }
        __syncthreads();

        // ---- phase 3: online softmax rows (wave 0, shfl-combined halves) ----
        if (wave == 0) {
            const int r2 = lane & 15;
            const int c0 = (lane >> 4) * 32;        // each lane scans 32 cols
            float mx = Sf[r2][c0];
#pragma unroll
            for (int n = 1; n < 32; ++n) mx = fmaxf(mx, Sf[r2][c0 + n]);
            mx = fmaxf(mx, __shfl_xor(mx, 16, 32));
            const float mold = mrow[r2];
            const float mnew = fmaxf(mold, mx);
            const float al   = __expf(mold - mnew);
            float sum = 0.f;
#pragma unroll
            for (int n = 0; n < 32; ++n) {
                const float p = __expf(Sf[r2][c0 + n] - mnew);
                sum += p;
                Plds[r2][c0 + n] = (__bf16)p;
            }
            sum += __shfl_xor(sum, 16, 32);
            if (lane < 16) {
                mrow[r2] = mnew;
                lrow[r2] = lrow[r2] * al + sum;
                arow[r2] = al;
            }
        }
        __syncthreads();

        // ---- phase 4: O = O*alpha + P @ V (16 WMMA) ----
        float alj[8];
#pragma unroll
        for (int j = 0; j < 8; ++j) alj[j] = arow[mh + j];

        AFrag pf[2];
        {
            const __bf16* pr = &Plds[rA][0];
            const int kb2 = (lane < 16) ? 0 : 8;
#pragma unroll
            for (int g = 0; g < 2; ++g) {
                pf[g].u[0] = *(const uint4*)(pr + g * 32 + kb2);
                pf[g].u[1] = *(const uint4*)(pr + g * 32 + kb2 + 16);
            }
        }
        const int kvB = (lane < 16) ? 0 : 8;
#pragma unroll
        for (int s = 0; s < 8; ++s) {
            const int e = wave * 128 + s * 16 + rA;
            const __bf16* vrow = Vb + (size_t)e * SEQ + kb + kvB;
            BFrag vf0, vf1;
            vf0.u[0] = *(const uint4*)(vrow);
            vf0.u[1] = *(const uint4*)(vrow + 16);
            vf1.u[0] = *(const uint4*)(vrow + 32);
            vf1.u[1] = *(const uint4*)(vrow + 48);
#pragma unroll
            for (int j = 0; j < 8; ++j) O[s][j] *= alj[j];
            O[s] = wmma_bf16(pf[0].v, vf0.v, O[s]);
            O[s] = wmma_bf16(pf[1].v, vf1.v, O[s]);
        }
        __syncthreads();
    }

    // ---- epilogue: normalize by l, store f32 coalesced ----
    float linv[8];
#pragma unroll
    for (int j = 0; j < 8; ++j) linv[j] = 1.0f / lrow[mh + j];
#pragma unroll
    for (int s = 0; s < 8; ++s) {
        const int e = wave * 128 + s * 16 + rA;
#pragma unroll
        for (int j = 0; j < 8; ++j) {
            const int qr = q0 + mh + j;
            Ob[(size_t)qr * EMBED + e] = O[s][j] * linv[j];
        }
    }
}

extern "C" void kernel_launch(void* const* d_in, const int* in_sizes, int n_in,
                              void* d_out, int out_size, void* d_ws, size_t ws_size,
                              hipStream_t stream) {
    const float* x    = (const float*)d_in[0];   // [8, 2048, 1024] f32
    const float* Wqkv = (const float*)d_in[1];   // [3072, 1024]    f32
    const float* bqkv = (const float*)d_in[2];   // [3072]          f32
    float* out = (float*)d_out;                  // [8, 2048, 1024] f32

    const size_t per = (size_t)BATCH * SEQ * EMBED;
    __bf16* Qws  = (__bf16*)d_ws;                // 32 MB
    __bf16* Kws  = Qws + per;                    // 32 MB
    __bf16* Vtws = Kws + per;                    // 32 MB (transposed V)

    dim3 g1((BATCH * SEQ) / 128, FOUT / 128);    // 128 x 24
    qkv_gemm_kernel<<<g1, 256, 0, stream>>>(x, Wqkv, bqkv, Qws, Kws, Vtws);

    dim3 g2(SEQ / 16, BATCH);                    // 128 x 8
    flash_attn_kernel<<<g2, 256, 0, stream>>>(Qws, Kws, Vtws, out);
}
